// SynthesisLayer_51384988729652
// MI455X (gfx1250) — matrix operand order repaired and checked by reading, not compiled, not run on hardware
//
#include <hip/hip_runtime.h>
#include <hip/hip_bf16.h>
#include <math.h>

// ---------------- problem constants ----------------
#define B_    4
#define CIN_  64
#define COUT_ 64
#define WDIM_ 512
#define RES_  512
#define KK_   3
#define LRELU_ 0.2f
#define GAIN_  1.41421356237309515f
#define CLAMP_ 256.0f

// main-kernel tiling
#define HTILE_ 4                 // output rows per block
#define WTILE_ 64                // output cols per block
#define XROWS_ (HTILE_ + 2)      // 6 halo rows
#define XCOLS_ (WTILE_ + 2)      // 66 halo cols
#define CPAD_  72                // padded cin stride (16B-aligned frag reads, odd-ish bank spread)

// packed-weight fragment workspace: [B][9][2][4][32][8] u32  (bf16 pairs)
#define FRAGW_U32_PER_B  (9 * 2 * 4 * 32 * 8)       // 18432
#define FRAGW_BYTES      (B_ * FRAGW_U32_PER_B * 4) // 294912
#define STYLES_OFF       FRAGW_BYTES                // fp32 styles [B][CIN] after fragW

#define LDS_W_BYTES  (FRAGW_U32_PER_B * 4)                    // 73728
#define LDS_X_BYTES  (XROWS_ * XCOLS_ * CPAD_ * 2)            // 57024
#define LDS_BYTES    (LDS_W_BYTES + LDS_X_BYTES)              // 130752 (< 320KB WGP LDS)

typedef __attribute__((ext_vector_type(16))) __bf16 v16bf;
typedef __attribute__((ext_vector_type(8)))  float  v8f;

// types matching the async-LDS builtin's parameters: v4i in AS1 / AS3
typedef int v4i __attribute__((vector_size(16)));
typedef __attribute__((address_space(1))) v4i as1_v4i;
typedef __attribute__((address_space(3))) v4i as3_v4i;

union Frag32B { uint4 q[2]; v16bf v; };

__device__ __forceinline__ unsigned short f2bf(float f) {
    unsigned int u = __float_as_uint(f);
    unsigned int r = u + 0x7FFFu + ((u >> 16) & 1u);   // round-to-nearest-even
    return (unsigned short)(r >> 16);
}

// ---------------------------------------------------------------------------
// Kernel 1: styles[b][cin] = (w[b,:] . affine_w[cin,:]) / sqrt(WDIM) + affine_b
// ---------------------------------------------------------------------------
__global__ void styles_kernel(const float* __restrict__ wlat,
                              const float* __restrict__ affw,
                              const float* __restrict__ affb,
                              float* __restrict__ styles) {
    const int b = blockIdx.x;
    const int cin = threadIdx.x;            // blockDim.x == CIN_
    float acc = 0.f;
    const float* wr = wlat + b * WDIM_;
    const float* ar = affw + cin * WDIM_;
    for (int d = 0; d < WDIM_; ++d) acc = fmaf(wr[d], ar[d], acc);
    styles[b * CIN_ + cin] = acc * 0.04419417382415922f + affb[cin]; // 1/sqrt(512)
}

// ---------------------------------------------------------------------------
// Kernel 2: modulate + demodulate, pack bf16 weights in WMMA A-fragment order.
// One wave32 per (b, cout).
// A-fragment (16-bit A 16x32, per ISA 7.12.2):
//   lane l: M = l%16 ; base_k = (l<16)?0:8
//   vgpr v<4 : K = base_k + 2v, +1 ; vgpr v>=4: K = 16 + base_k + 2(v-4), +1
// ---------------------------------------------------------------------------
__global__ void modw_kernel(const float* __restrict__ weight,
                            const float* __restrict__ styles,
                            unsigned int* __restrict__ fragW) {
    const int bc = blockIdx.x;
    const int b = bc >> 6;
    const int cout = bc & 63;
    const int lane = threadIdx.x;           // 32 threads

    // pass 1: sum of squares of modulated weights for this (b, cout)
    float ssq = 0.f;
    for (int e = lane; e < CIN_ * 9; e += 32) {
        const int cin = e / 9, kidx = e % 9;
        const float v = weight[(cout * CIN_ + cin) * 9 + kidx] * styles[b * CIN_ + cin];
        ssq = fmaf(v, v, ssq);
    }
    #pragma unroll
    for (int off = 16; off > 0; off >>= 1) ssq += __shfl_xor(ssq, off, 32);
    const float demod = rsqrtf(ssq + 1e-8f);

    // pass 2: pack. Each lane owns one (kb, half, v) slot, iterates kidx.
    const int m  = cout & 15;
    const int cg = cout >> 4;
    const int v    = lane & 7;
    const int hsel = (lane >> 3) & 1;
    const int kb   = lane >> 4;
    const int fl    = m + hsel * 16;
    const int basek = hsel * 8;
    const int K0 = (v < 4) ? (basek + 2 * v) : (16 + basek + 2 * (v - 4));
    const int cin0 = kb * 32 + K0;
    const float s0 = styles[b * CIN_ + cin0];
    const float s1 = styles[b * CIN_ + cin0 + 1];
    #pragma unroll
    for (int kidx = 0; kidx < 9; ++kidx) {
        const float w0 = weight[(cout * CIN_ + cin0)     * 9 + kidx] * s0 * demod;
        const float w1 = weight[(cout * CIN_ + cin0 + 1) * 9 + kidx] * s1 * demod;
        const unsigned int packed = (unsigned int)f2bf(w0) | ((unsigned int)f2bf(w1) << 16);
        fragW[((((b * 9 + kidx) * 2 + kb) * 4 + cg) * 32 + fl) * 8 + v] = packed;
    }
}

// ---------------------------------------------------------------------------
// Kernel 3: implicit-GEMM 3x3 conv with bf16 WMMA + fused epilogue.
// Block: 256 thr (8 waves). Computes [COUT=64] x [HTILE=4 rows] x [WTILE=64 px].
// ---------------------------------------------------------------------------
__global__ void __launch_bounds__(256)
conv_kernel(const float* __restrict__ x,
            const float* __restrict__ noise,
            const float* __restrict__ bias,
            const float* __restrict__ nstr,
            const unsigned int* __restrict__ fragW,
            float* __restrict__ out) {
    extern __shared__ char smem[];
    char*           sW = smem;                                   // packed A fragments
    unsigned short* sX = (unsigned short*)(smem + LDS_W_BYTES);  // [6][66][CPAD_] bf16

    const int b  = blockIdx.z;
    const int h0 = blockIdx.y * HTILE_;
    const int w0 = blockIdx.x * WTILE_;

    // ---- stage per-batch packed weights (raw bytes, already bf16) ----
    const uint4* gW4 = (const uint4*)(fragW + (size_t)b * FRAGW_U32_PER_B);
#if __has_builtin(__builtin_amdgcn_global_load_async_to_lds_b128) && \
    __has_builtin(__builtin_amdgcn_s_wait_asynccnt)
    for (int i = threadIdx.x; i < LDS_W_BYTES / 16; i += 256) {
        __builtin_amdgcn_global_load_async_to_lds_b128(
            (as1_v4i*)(gW4 + i), (as3_v4i*)(sW + i * 16), 0, 0);
    }
    __builtin_amdgcn_s_wait_asynccnt(0);
#else
    {
        uint4* sW4 = (uint4*)sW;
        for (int i = threadIdx.x; i < LDS_W_BYTES / 16; i += 256) sW4[i] = gW4[i];
    }
#endif

    // ---- stage input halo tile, fp32 -> bf16, channel-contiguous layout ----
    for (int e = threadIdx.x; e < XROWS_ * XCOLS_ * CIN_; e += 256) {
        const int p  = e % XCOLS_;          // consecutive threads -> consecutive w (coalesced)
        const int t2 = e / XCOLS_;
        const int r  = t2 % XROWS_;
        const int c  = t2 / XROWS_;
        const int h = h0 - 1 + r;
        const int w = w0 - 1 + p;
        float v = 0.f;
        if (h >= 0 && h < RES_ && w >= 0 && w < RES_)
            v = x[(((size_t)b * CIN_ + c) * RES_ + h) * RES_ + w];
        sX[(r * XCOLS_ + p) * CPAD_ + c] = f2bf(v);
    }
    __syncthreads();

    // ---- per-wave tile assignment ----
    const int wave = threadIdx.x >> 5;      // 0..7
    const int lane = threadIdx.x & 31;
    const int cout_g  = wave >> 1;          // 0..3  (16 couts)
    const int pg_base = (wave & 1) * 2;     // px groups {0,1} or {2,3}
    const int nIdx = lane & 15;
    const int hsel = lane >> 4;

    v8f acc[8];                             // [hr][pgi] tiles
    #pragma unroll
    for (int i = 0; i < 8; ++i)
        #pragma unroll
        for (int j = 0; j < 8; ++j) acc[i][j] = 0.f;

    // ---- main WMMA loop: 9 taps x 2 K-blocks of 32 cin ----
    for (int kidx = 0; kidx < 9; ++kidx) {
        const int kh = kidx / 3, kw = kidx % 3;
        #pragma unroll
        for (int kb = 0; kb < 2; ++kb) {
            Frag32B a;
            const char* aoff = sW + (((kidx * 2 + kb) * 4 + cout_g) * 1024 + lane * 32);
            a.q[0] = *(const uint4*)aoff;
            a.q[1] = *(const uint4*)(aoff + 16);
            // B-fragment K base: lanes 0-15 -> K 0..7,16..23, lanes 16-31 -> K 8..15,24..31
            const int cbase = kb * 32 + hsel * 8;
            #pragma unroll
            for (int hr = 0; hr < HTILE_; ++hr) {
                const int r = hr + kh;
                #pragma unroll
                for (int pgi = 0; pgi < 2; ++pgi) {
                    const int p = (pg_base + pgi) * 16 + nIdx + kw;
                    Frag32B bf;
                    const char* boff = (const char*)sX + ((r * XCOLS_ + p) * CPAD_ + cbase) * 2;
                    bf.q[0] = *(const uint4*)boff;
                    bf.q[1] = *(const uint4*)(boff + 32);
                    acc[hr * 2 + pgi] = __builtin_amdgcn_wmma_f32_16x16x32_bf16(
                        false, a.v, false, bf.v, (short)0, acc[hr * 2 + pgi],
                        false, false);
                }
            }
        }
    }

    // ---- fused epilogue: noise + bias + lrelu*gain + clamp, fp32 store ----
    const float ns = nstr[0];
    float bv[8];
    #pragma unroll
    for (int rr = 0; rr < 8; ++rr) bv[rr] = bias[cout_g * 16 + hsel * 8 + rr];

    #pragma unroll
    for (int hr = 0; hr < HTILE_; ++hr) {
        const int h = h0 + hr;
        #pragma unroll
        for (int pgi = 0; pgi < 2; ++pgi) {
            const int w = w0 + (pg_base + pgi) * 16 + nIdx;
            const float nz = noise[((size_t)b * RES_ + h) * RES_ + w] * ns;
            const v8f av = acc[hr * 2 + pgi];
            #pragma unroll
            for (int rr = 0; rr < 8; ++rr) {
                const int cout = cout_g * 16 + hsel * 8 + rr;   // C/D layout: M = rr + 8*hsel
                float v = av[rr] + nz + bv[rr];
                v = (v >= 0.f ? v : v * LRELU_) * GAIN_;
                v = fminf(fmaxf(v, -CLAMP_), CLAMP_);
                out[(((size_t)b * COUT_ + cout) * RES_ + h) * RES_ + w] = v;
            }
        }
    }
}

// ---------------------------------------------------------------------------
extern "C" void kernel_launch(void* const* d_in, const int* in_sizes, int n_in,
                              void* d_out, int out_size, void* d_ws, size_t ws_size,
                              hipStream_t stream) {
    const float* x      = (const float*)d_in[0];
    const float* wlat   = (const float*)d_in[1];
    const float* noise  = (const float*)d_in[2];
    const float* weight = (const float*)d_in[3];
    const float* affw   = (const float*)d_in[4];
    const float* affb   = (const float*)d_in[5];
    const float* nstr   = (const float*)d_in[6];
    const float* bias   = (const float*)d_in[7];
    float* out = (float*)d_out;

    unsigned int* fragW  = (unsigned int*)d_ws;                    // 294912 B
    float*        styles = (float*)((char*)d_ws + STYLES_OFF);     // + 1024 B

    styles_kernel<<<dim3(B_), dim3(CIN_), 0, stream>>>(wlat, affw, affb, styles);
    modw_kernel<<<dim3(B_ * COUT_), dim3(32), 0, stream>>>(weight, styles, fragW);

    dim3 grid(RES_ / WTILE_, RES_ / HTILE_, B_);   // (8, 128, 4)
    conv_kernel<<<grid, dim3(256), LDS_BYTES, stream>>>(x, noise, bias, nstr, fragW, out);
}
